// MADEnew2d_36936718746166
// MI455X (gfx1250) — matrix operand order repaired and checked by reading, not compiled, not run on hardware
//
#include <hip/hip_runtime.h>
#include <hip/hip_bf16.h>

typedef __attribute__((ext_vector_type(16))) _Float16 v16h;
typedef __attribute__((ext_vector_type(8)))  _Float16 v8h;
typedef __attribute__((ext_vector_type(4)))  _Float16 v4h;
typedef __attribute__((ext_vector_type(8)))  float    v8f;

#define BATCH 512
#define NPIX  1024   // 32*32
#define HID   4096   // WID*N

// ---------------------------------------------------------------------------
// Build the quad-tree fold permutation (matches _fold_perm(L=32)).
// Tiny, deterministic, runs once per call on one thread.
// ---------------------------------------------------------------------------
__global__ void build_perm_kernel(int* __restrict__ perm) {
    if (blockIdx.x != 0 || threadIdx.x != 0) return;
    int idx = 0;
    perm[idx++] = 0;
    for (int i = 4; i >= 0; --i) {
        const int s = 1 << i, t = s << 1;
        for (int r = s; r < 32; r += t)
            for (int c = s; c < 32; c += t) perm[idx++] = r * 32 + c;
        for (int r = s; r < 32; r += t)
            for (int c = 0; c < 32; c += t) perm[idx++] = r * 32 + c;
        for (int r = 0; r < 32; r += t)
            for (int c = s; c < 32; c += t) perm[idx++] = r * 32 + c;
    }
}

// ---------------------------------------------------------------------------
// fp32 -> f16 weight conversion, 4 elements / thread (vectorized 16B loads).
// ---------------------------------------------------------------------------
__global__ void cvt_f32_to_f16_x4(const float4* __restrict__ src,
                                  v4h* __restrict__ dst, int n4) {
    int i = blockIdx.x * blockDim.x + threadIdx.x;
    if (i < n4) {
        float4 f = src[i];
        v4h h;
        h.x = (_Float16)f.x; h.y = (_Float16)f.y;
        h.z = (_Float16)f.z; h.w = (_Float16)f.w;
        dst[i] = h;
    }
}

// ---------------------------------------------------------------------------
// act0[b][i] = (f16) x[b][perm[i]]   (input gather + downconvert)
// ---------------------------------------------------------------------------
__global__ void permute_x_kernel(const float* __restrict__ x,
                                 const int* __restrict__ perm,
                                 _Float16* __restrict__ act0) {
    int i = blockIdx.x * blockDim.x + threadIdx.x;   // over BATCH*NPIX
    if (i < BATCH * NPIX) {
        int b = i >> 10, j = i & 1023;
        act0[i] = (_Float16)x[(size_t)b * NPIX + perm[j]];
    }
}

// ---------------------------------------------------------------------------
// WMMA GEMM:  H[M=512, O] = X[512, K] @ W[O, K]^T + bias
// Block: 256 threads = 8 waves laid out 2(M) x 4(N).
// Wave tile: 32x64 = 2x4 WMMA 16x16 frags; block tile 64 x 256.
// A/B fragments loaded directly from global (weights+acts are L2-resident:
// f16 weights total 80MB < 192MB L2).
// MODE 0: epilogue = bias + PReLU(slope)  -> f16 activations for next layer
// MODE 1: epilogue = bias + sigmoid, scatter through perm into fp32 d_out,
//         with pixel 0 forced to 0.5 (the xm/xb mask in the reference).
// ---------------------------------------------------------------------------
template <int MODE>
__global__ __launch_bounds__(256)
void gemm_wmma_kernel(const _Float16* __restrict__ X,
                      const _Float16* __restrict__ W,
                      const float* __restrict__ bias,
                      const float* __restrict__ slope,
                      _Float16* __restrict__ Y16,
                      float* __restrict__ Yout,
                      const int* __restrict__ perm,
                      int K, int O) {
    const int lane = threadIdx.x & 31;
    const int wave = threadIdx.x >> 5;
    const int wm = wave & 1;        // wave position in M
    const int wn = wave >> 1;       // wave position in N
    const int m0 = blockIdx.y * 64 + wm * 32;
    const int n0 = blockIdx.x * 256 + wn * 64;
    const int lr = lane & 15;       // row-within-frag selector
    const int lh = lane >> 4;       // half selector

    v8f acc[2][4];
#pragma unroll
    for (int im = 0; im < 2; ++im)
#pragma unroll
        for (int in = 0; in < 4; ++in)
            acc[im][in] = (v8f){0.f, 0.f, 0.f, 0.f, 0.f, 0.f, 0.f, 0.f};

    union Frag { v16h v; v8h h[2]; };
    const int aoff = lh ? 8 : 0;    // A frag: lanes>=16 start at K+8

    for (int kb = 0; kb < K; kb += 32) {
        Frag A[2], B[4];
        // A: 16x32 f16. lane<16: row=lr, K = kb+{0..7,16..23}
        //               lane>=16: row=lr, K = kb+{8..15,24..31}
#pragma unroll
        for (int im = 0; im < 2; ++im) {
            const _Float16* p = X + (size_t)(m0 + im * 16 + lr) * K + kb + aoff;
            A[im].h[0] = *(const v8h*)(p);
            A[im].h[1] = *(const v8h*)(p + 16);
        }
        // B: 32x16 f16 (B[k][n] = W[n][k]). lane holds col n=lr,
        // K = kb + lh*8 + {0..7,16..23} (same order as A).
#pragma unroll
        for (int in = 0; in < 4; ++in) {
            const _Float16* p = W + (size_t)(n0 + in * 16 + lr) * K + kb + lh * 8;
            B[in].h[0] = *(const v8h*)(p);
            B[in].h[1] = *(const v8h*)(p + 16);
        }
#pragma unroll
        for (int im = 0; im < 2; ++im)
#pragma unroll
            for (int in = 0; in < 4; ++in)
                acc[im][in] = __builtin_amdgcn_wmma_f32_16x16x32_f16(
                    false, A[im].v, false, B[in].v,
                    (short)0, acc[im][in], false, false);
    }

    // Epilogue. C/D layout: VGPR r -> row (lh*8 + r), col lr (within 16x16).
#pragma unroll
    for (int im = 0; im < 2; ++im) {
        const int mbase = m0 + im * 16 + lh * 8;
#pragma unroll
        for (int in = 0; in < 4; ++in) {
            const int col = n0 + in * 16 + lr;
            const float bb = bias[col];
            if (MODE == 0) {
                const float sl = slope[col];
#pragma unroll
                for (int r = 0; r < 8; ++r) {
                    float h = acc[im][in][r] + bb;
                    h = (h >= 0.f) ? h : sl * h;
                    Y16[(size_t)(mbase + r) * O + col] = (_Float16)h;
                }
            } else {
                const int p = perm[col];   // out[:, inv] == scatter via perm
#pragma unroll
                for (int r = 0; r < 8; ++r) {
                    float h = acc[im][in][r] + bb;
                    float s = 1.f / (1.f + __expf(-h));
                    Yout[(size_t)(mbase + r) * NPIX + p] = (p == 0) ? 0.5f : s;
                }
            }
        }
    }
}

// ---------------------------------------------------------------------------
extern "C" void kernel_launch(void* const* d_in, const int* in_sizes, int n_in,
                              void* d_out, int out_size, void* d_ws, size_t ws_size,
                              hipStream_t stream) {
    (void)in_sizes; (void)n_in; (void)out_size; (void)ws_size;

    const float* x  = (const float*)d_in[0];
    const float* w0 = (const float*)d_in[1];
    const float* b0 = (const float*)d_in[2];
    const float* a1 = (const float*)d_in[3];
    const float* w1 = (const float*)d_in[4];
    const float* b1 = (const float*)d_in[5];
    const float* a2 = (const float*)d_in[6];
    const float* w2 = (const float*)d_in[7];
    const float* b2 = (const float*)d_in[8];
    const float* a3 = (const float*)d_in[9];
    const float* w3 = (const float*)d_in[10];
    const float* b3 = (const float*)d_in[11];
    float* out = (float*)d_out;

    // ---- workspace carve-up (bytes) ----
    char* ws = (char*)d_ws;
    size_t o = 0;
    int* perm = (int*)(ws + o);              o += 4096;
    _Float16* w0f  = (_Float16*)(ws + o);    o += (size_t)HID  * NPIX * 2;  // 8 MB
    _Float16* w1f  = (_Float16*)(ws + o);    o += (size_t)HID  * HID  * 2;  // 32 MB
    _Float16* w2f  = (_Float16*)(ws + o);    o += (size_t)HID  * HID  * 2;  // 32 MB
    _Float16* w3f  = (_Float16*)(ws + o);    o += (size_t)NPIX * HID  * 2;  // 8 MB
    _Float16* act0 = (_Float16*)(ws + o);    o += (size_t)BATCH * NPIX * 2; // 1 MB
    _Float16* act1 = (_Float16*)(ws + o);    o += (size_t)BATCH * HID * 2;  // 4 MB
    _Float16* act2 = (_Float16*)(ws + o);    o += (size_t)BATCH * HID * 2;  // 4 MB
    _Float16* act3 = (_Float16*)(ws + o);    o += (size_t)BATCH * HID * 2;  // 4 MB

    // 1) permutation table
    build_perm_kernel<<<1, 1, 0, stream>>>(perm);

    // 2) weight downconversion fp32 -> f16
    auto cvt = [&](const float* s, _Float16* d, size_t n) {
        int n4 = (int)(n / 4);
        cvt_f32_to_f16_x4<<<(n4 + 255) / 256, 256, 0, stream>>>(
            (const float4*)s, (v4h*)d, n4);
    };
    cvt(w0, w0f, (size_t)HID * NPIX);
    cvt(w1, w1f, (size_t)HID * HID);
    cvt(w2, w2f, (size_t)HID * HID);
    cvt(w3, w3f, (size_t)NPIX * HID);

    // 3) input gather + downconvert
    permute_x_kernel<<<(BATCH * NPIX) / 256, 256, 0, stream>>>(x, perm, act0);

    // 4) four WMMA GEMM layers
    dim3 blk(256);
    dim3 gh(HID / 256, BATCH / 64);   // (16, 8) for O=4096
    dim3 go(NPIX / 256, BATCH / 64);  // (4, 8)  for O=1024

    gemm_wmma_kernel<0><<<gh, blk, 0, stream>>>(act0, w0f, b0, a1, act1, nullptr,
                                                nullptr, NPIX, HID);
    gemm_wmma_kernel<0><<<gh, blk, 0, stream>>>(act1, w1f, b1, a2, act2, nullptr,
                                                nullptr, HID, HID);
    gemm_wmma_kernel<0><<<gh, blk, 0, stream>>>(act2, w2f, b2, a3, act3, nullptr,
                                                nullptr, HID, HID);
    gemm_wmma_kernel<1><<<go, blk, 0, stream>>>(act3, w3f, b3, nullptr, nullptr, out,
                                                perm, HID, NPIX);
}
